// GatedDeltaNetPoST_42125039239461
// MI455X (gfx1250) — matrix operand order repaired and checked by reading, not compiled, not run on hardware
//
#include <hip/hip_runtime.h>

// ---------------------------------------------------------------------------
// Gated DeltaNet forward for MI455X (gfx1250, wave32, WMMA + TDM async tensor)
// B=2, T=4096, D=1024, H=8, K=128, V=256, CONV=4
// ---------------------------------------------------------------------------

typedef __attribute__((ext_vector_type(16))) __bf16        v16bf;
typedef __attribute__((ext_vector_type(8)))  __bf16        v8bf;
typedef __attribute__((ext_vector_type(8)))  float         v8f;
typedef __attribute__((ext_vector_type(4)))  unsigned int  v4u;
typedef __attribute__((ext_vector_type(4)))  int           v4i;
typedef __attribute__((ext_vector_type(8)))  int           v8i;

static constexpr int Bc = 2, Tc = 4096, Dc = 1024, Hc = 8, Kc = 128, Vc = 256;
static constexpr int BT = Bc * Tc;   // 8192 rows

// LDS B-panel geometry: 32 K-rows x 64 N-cols, padded to 72 elems/row so the
// row stride is 144B (16B aligned for ds_load_b128, 2-way max bank conflict).
static constexpr int BP_W = 64, BP_H = 32, BP_STRIDE = 72;

__device__ __forceinline__ __bf16 f2bf(float f) {
  unsigned u = __float_as_uint(f);
  unsigned r = (u + 0x7FFFu + ((u >> 16) & 1u)) >> 16;   // round-to-nearest-even
  return __builtin_bit_cast(__bf16, (unsigned short)r);
}

// ---------------- TDM: async 2D bf16 tile load, global -> LDS --------------
// D# group0: count=1, lds_addr, global_addr(tile start), type=2
// D# group1: data_size=1(2B), pad_enable, pad_interval=32 DW (code 4),
//            pad_amount=4 DW (code 3), tensor/tile dims + dim0 stride.
#if __has_builtin(__builtin_amdgcn_tensor_load_to_lds)
#define GDN_HAVE_TDM 1
__device__ __forceinline__ void tdm_load_b_panel(__bf16* ldsdst,
                                                 const __bf16* gsrc,
                                                 int row_stride_elems) {
  const unsigned long long ga = (unsigned long long)(size_t)gsrc;
  v4u g0;
  g0[0] = 1u;                                   // count=1 (valid user D#)
  g0[1] = (unsigned)(size_t)ldsdst;             // LDS byte address
  g0[2] = (unsigned)(ga & 0xFFFFFFFFu);         // global_addr[31:0]
  g0[3] = (unsigned)((ga >> 32) & 0x01FFFFFFu)  // global_addr[56:32]
        | 0x80000000u;                          // type=2 ("image")
  v8i g1;
  g1[0] = (1 << 16)                             // data_size: 2 bytes
        | (1 << 20)                             // pad_enable
        | (4 << 22)                             // pad_interval: 32 DWORDs
        | (3 << 25);                            // pad_amount: 4 DWORDs (8 elems)
  g1[1] = (int)(((unsigned)row_stride_elems & 0xFFFFu) << 16);     // dim0[15:0]
  g1[2] = (int)((((unsigned)row_stride_elems >> 16) & 0xFFFFu)     // dim0[31:16]
        | ((unsigned)BP_H << 16));                                 // dim1[15:0]
  g1[3] = (int)((unsigned)BP_W << 16);          // dim1[31:16]=0 | tile_dim0=64
  g1[4] = BP_H;                                 // tile_dim1=32 | tile_dim2=0
  g1[5] = row_stride_elems;                     // tensor_dim0_stride[31:0]
  g1[6] = 0;                                    // stride0[47:32] | stride1[15:0]
  g1[7] = 0;                                    // stride1[47:16]
  v4i z4 = {0, 0, 0, 0};
#if defined(__clang_major__) && (__clang_major__ >= 23)
  v8i z8 = {0, 0, 0, 0, 0, 0, 0, 0};
  __builtin_amdgcn_tensor_load_to_lds(g0, g1, z4, z4, z8, 0);
#else
  __builtin_amdgcn_tensor_load_to_lds(g0, g1, z4, z4, 0);
#endif
}
#else
#define GDN_HAVE_TDM 0
#endif

// ------------------------- fp32 -> bf16 convert ----------------------------
__global__ void cvt_f32_bf16(const float* __restrict__ x, __bf16* __restrict__ y,
                             size_t n) {
  size_t i = (size_t)blockIdx.x * blockDim.x + threadIdx.x;
  size_t stride = (size_t)gridDim.x * blockDim.x;
  for (; i < n; i += stride) y[i] = f2bf(x[i]);
}

// ---------------- WMMA bf16 GEMM: C[M,N] = A[M,Kd] * Bw[Kd,N] --------------
// 4-wave workgroup computes a 64x64 tile. B panel (32x64) staged into LDS by
// the Tensor Data Mover (double buffered, TENSORcnt), shared by all 4 waves.
// A frags (16x32/wave) are register-direct global loads reused over 4 N-tiles.
// A 16x32 layout: lane%16 = M row, two 8-elem K runs at hlf*8 and 16+hlf*8.
// B 32x16 layout: lane%16 = N col, two 8-elem K runs at hlf*8 and 16+hlf*8 (from LDS).
__global__ __launch_bounds__(128)
void wmma_gemm_bf16(const __bf16* __restrict__ A, const __bf16* __restrict__ Bw,
                    float* __restrict__ C, int M, int N, int Kd) {
  __shared__ __bf16 bpanel[2][BP_H][BP_STRIDE];   // 2 x 4.5 KB
  const int tid  = threadIdx.x;
  const int lane = tid & 31;
  const int wv   = tid >> 5;
  const int l16  = lane & 15;
  const int hlf  = lane >> 4;
  const int m0 = blockIdx.x * 64 + wv * 16;
  const int n0 = blockIdx.y * 64;

  const __bf16* arow = A + (size_t)(m0 + l16) * Kd + hlf * 8;
  v8f acc[4] = {};

#if GDN_HAVE_TDM
  if (tid < 32)   // wave 0 drives the TDM
    tdm_load_b_panel(&bpanel[0][0][0], Bw + (size_t)0 * N + n0, N);
#endif

  int cur = 0;
  for (int kk = 0; kk < Kd; kk += 32) {
#if GDN_HAVE_TDM
    if (tid < 32) {
      if (kk + 32 < Kd) {
        tdm_load_b_panel(&bpanel[cur ^ 1][0][0],
                         Bw + (size_t)(kk + 32) * N + n0, N);
        __builtin_amdgcn_s_wait_tensorcnt(1);   // current panel complete
      } else {
        __builtin_amdgcn_s_wait_tensorcnt(0);
      }
    }
#else
    // fallback: cooperative global->LDS copy of the 32x64 panel
    for (int e = tid; e < BP_H * (BP_W / 8); e += 128) {
      const int r = e / (BP_W / 8), c = (e % (BP_W / 8)) * 8;
      *(v8bf*)&bpanel[cur][r][c] =
          *(const v8bf*)(Bw + (size_t)(kk + r) * N + n0 + c);
    }
#endif
    __syncthreads();

    v8bf alo = *(const v8bf*)(arow + kk);
    v8bf ahi = *(const v8bf*)(arow + kk + 16);
    v16bf af = __builtin_shufflevector(alo, ahi,
                0, 1, 2, 3, 4, 5, 6, 7, 8, 9, 10, 11, 12, 13, 14, 15);
    __builtin_prefetch(arow + kk + 64, 0, 1);   // next A fragment

    const __bf16* bcol = &bpanel[cur][0][l16];
#pragma unroll
    for (int nt = 0; nt < 4; ++nt) {
      v16bf bfr;
#pragma unroll
      for (int s = 0; s < 8; ++s) {
        bfr[s]     = bcol[(hlf * 8 + s) * BP_STRIDE + nt * 16];
        bfr[8 + s] = bcol[(16 + hlf * 8 + s) * BP_STRIDE + nt * 16];
      }
      acc[nt] = __builtin_amdgcn_wmma_f32_16x16x32_bf16(
          /*neg_a=*/false, af, /*neg_b=*/false, bfr,
          /*c_mod=*/(short)0, acc[nt], /*reuse_a=*/false, /*reuse_b=*/false);
    }
    __syncthreads();      // panel consumed before next TDM overwrite
    cur ^= 1;
  }

#pragma unroll
  for (int nt = 0; nt < 4; ++nt) {
    const int col = n0 + nt * 16 + l16;
#pragma unroll
    for (int r = 0; r < 8; ++r) {
      const int row = m0 + hlf * 8 + r;
      C[(size_t)row * N + col] = acc[nt][r];
    }
  }
}

// ------- causal depthwise conv(4) + SiLU + per-head L2 norm (q,k) ----------
// one wave per (b,t,h); lane handles 4 of the 128 head channels
__global__ __launch_bounds__(32)
void conv_silu_l2norm_qk(const float* __restrict__ p, const float* __restrict__ w,
                         float* __restrict__ out, float scale) {
  const int gid  = blockIdx.x;          // (b*T + t)*H + h
  const int h    = gid & (Hc - 1);
  const int bt   = gid >> 3;
  const int t    = bt & (Tc - 1);
  const int lane = threadIdx.x;

  float y[4];
  float ss = 0.f;
#pragma unroll
  for (int j = 0; j < 4; ++j) {
    const int c = h * Kc + lane * 4 + j;
    float acc = 0.f;
#pragma unroll
    for (int i = 0; i < 4; ++i) {
      const int tt = t - 3 + i;
      if (tt >= 0)
        acc += p[(size_t)(bt - 3 + i) * (Hc * Kc) + c] * w[c * 4 + i];
    }
    const float s = acc / (1.f + __expf(-acc));   // SiLU
    y[j] = s;
    ss += s * s;
  }
#pragma unroll
  for (int off = 16; off > 0; off >>= 1) ss += __shfl_xor(ss, off, 32);
  const float rn = rsqrtf(ss + 1e-6f) * scale;
#pragma unroll
  for (int j = 0; j < 4; ++j)
    out[(size_t)bt * (Hc * Kc) + h * Kc + lane * 4 + j] = y[j] * rn;
}

// ----------------- causal depthwise conv(4) + SiLU (v) ---------------------
__global__ void conv_silu_v(const float* __restrict__ p, const float* __restrict__ w,
                            float* __restrict__ out) {
  const int C = Hc * Vc;
  const size_t n = (size_t)BT * C;
  size_t idx = (size_t)blockIdx.x * blockDim.x + threadIdx.x;
  if (idx >= n) return;
  const int c  = (int)(idx % C);
  const size_t bt = idx / C;
  const int t = (int)(bt & (Tc - 1));
  float acc = 0.f;
#pragma unroll
  for (int i = 0; i < 4; ++i) {
    const int tt = t - 3 + i;
    if (tt >= 0) acc += p[(bt - 3 + i) * C + c] * w[c * 4 + i];
  }
  out[idx] = acc / (1.f + __expf(-acc));
}

// ------------- beta = sigmoid(x@Wb), g = -exp(A)*softplus(x@Wa+b) ----------
__global__ __launch_bounds__(32)
void beta_g_kernel(const float* __restrict__ x, const float* __restrict__ Wb,
                   const float* __restrict__ Wa, const float* __restrict__ A_log,
                   const float* __restrict__ dt_bias,
                   float* __restrict__ beta, float* __restrict__ g) {
  const int bt   = blockIdx.x;
  const int lane = threadIdx.x;
  const float* xr = x + (size_t)bt * Dc;
  float ab[8] = {}, aa[8] = {};
  for (int d = lane; d < Dc; d += 32) {
    const float xv = xr[d];
#pragma unroll
    for (int h = 0; h < 8; ++h) {
      ab[h] += xv * Wb[d * Hc + h];
      aa[h] += xv * Wa[d * Hc + h];
    }
  }
#pragma unroll
  for (int h = 0; h < 8; ++h) {
#pragma unroll
    for (int off = 16; off > 0; off >>= 1) {
      ab[h] += __shfl_xor(ab[h], off, 32);
      aa[h] += __shfl_xor(aa[h], off, 32);
    }
  }
  if (lane < 8) {
    float bsum = ab[0], asum = aa[0];
#pragma unroll
    for (int h = 1; h < 8; ++h)
      if (lane == h) { bsum = ab[h]; asum = aa[h]; }
    beta[(size_t)bt * Hc + lane] = 1.f / (1.f + __expf(-bsum));
    const float z  = asum + dt_bias[lane];
    const float sp = (z > 20.f) ? z : log1pf(__expf(z));
    g[(size_t)bt * Hc + lane] = -__expf(A_log[lane]) * sp;
  }
}

// ------------------------- gated delta-rule scan ---------------------------
// One 1024-thread block per (b,h). State S[K=128,V=256] in LDS (128 KB).
// Column v is split over 4 lanes (interleaved k-slices, rows kk%4==ks);
// bank swizzle (v + 16*row)&255 makes the 4 slices x 8 columns of each wave
// hit 32 distinct banks. RMSNorm fused, bf16 output for the final WMMA GEMM.
__global__ __launch_bounds__(1024)
void gdn_scan_kernel(const float* __restrict__ qn, const float* __restrict__ kn,
                     const float* __restrict__ vn, const float* __restrict__ g,
                     const float* __restrict__ beta, const float* __restrict__ norm_w,
                     __bf16* __restrict__ ob) {
  extern __shared__ float smem[];
  float* S   = smem;               // Kc*Vc swizzled
  float* sq  = S + Kc * Vc;        // Kc
  float* sk  = sq + Kc;            // Kc
  float* red = sk + Kc;            // Vc
  const int b   = blockIdx.x >> 3;
  const int h   = blockIdx.x & 7;
  const int tid = threadIdx.x;     // 0..1023
  const int v   = tid >> 2;        // column 0..255
  const int ks  = tid & 3;         // k-slice

  for (int i = tid; i < Kc * Vc; i += 1024) S[i] = 0.f;
  __syncthreads();
  const float nwv = norm_w[v];

  int sidx[32];
#pragma unroll
  for (int i = 0; i < 32; ++i) {
    const int row = i * 4 + ks;
    sidx[i] = row * Vc + ((v + 16 * row) & (Vc - 1));
  }

  for (int t = 0; t < Tc; ++t) {
    const size_t bt = (size_t)b * Tc + t;
    if (tid < Kc)            sq[tid]       = qn[bt * (Hc * Kc) + h * Kc + tid];
    else if (tid < 2 * Kc)   sk[tid - Kc]  = kn[bt * (Hc * Kc) + h * Kc + tid - Kc];
    const float vt  = vn[bt * (Hc * Vc) + h * Vc + v];
    const float gt  = g[bt * Hc + h];
    const float btv = beta[bt * Hc + h];
    __syncthreads();

    const float dec = __expf(gt);
    // phase 1: S *= exp(g); acc = (k . S_col)
    float acc = 0.f;
#pragma unroll
    for (int i = 0; i < 32; ++i) {
      const float s = S[sidx[i]] * dec;
      S[sidx[i]] = s;
      acc = fmaf(sk[i * 4 + ks], s, acc);
    }
    acc += __shfl_xor(acc, 1, 32);
    acc += __shfl_xor(acc, 2, 32);
    const float vnew = btv * (vt - acc);

    // phase 2: S += k (x) vnew; o = (q . S_col)
    float o = 0.f;
#pragma unroll
    for (int i = 0; i < 32; ++i) {
      const float s = fmaf(sk[i * 4 + ks], vnew, S[sidx[i]]);
      S[sidx[i]] = s;
      o = fmaf(sq[i * 4 + ks], s, o);
    }
    o += __shfl_xor(o, 1, 32);
    o += __shfl_xor(o, 2, 32);

    // fused per-head RMSNorm over V
    if (ks == 0) red[v] = o * o;
    __syncthreads();
    for (int off = 128; off > 0; off >>= 1) {
      if (tid < off) red[tid] += red[tid + off];
      __syncthreads();
    }
    const float rn = rsqrtf(red[0] * (1.f / Vc) + 1e-5f);
    if (ks == 0)
      ob[bt * (Hc * Vc) + h * Vc + v] = f2bf(o * rn * nwv);
    __syncthreads();
  }
}

// ---------------------------------------------------------------------------
extern "C" void kernel_launch(void* const* d_in, const int* in_sizes, int n_in,
                              void* d_out, int out_size, void* d_ws, size_t ws_size,
                              hipStream_t stream) {
  const float* x      = (const float*)d_in[0];
  const float* Wq     = (const float*)d_in[1];
  const float* Wk     = (const float*)d_in[2];
  const float* Wv     = (const float*)d_in[3];
  const float* Wb     = (const float*)d_in[4];
  const float* Wa     = (const float*)d_in[5];
  const float* A_log  = (const float*)d_in[6];
  const float* dtb    = (const float*)d_in[7];
  const float* cqw    = (const float*)d_in[8];
  const float* ckw    = (const float*)d_in[9];
  const float* cvw    = (const float*)d_in[10];
  const float* norm_w = (const float*)d_in[11];
  const float* Wo     = (const float*)d_in[12];
  float* out = (float*)d_out;

  char* ws = (char*)d_ws;
  size_t off = 0;
  auto alloc = [&](size_t bytes) -> void* {
    void* p = ws + off;
    off += (bytes + 255) & ~(size_t)255;
    return p;
  };
  __bf16* xb  = (__bf16*)alloc((size_t)BT * Dc * 2);           // x in bf16
  __bf16* wqb = (__bf16*)alloc((size_t)Dc * (Hc * Kc) * 2);
  __bf16* wkb = (__bf16*)alloc((size_t)Dc * (Hc * Kc) * 2);
  __bf16* wvb = (__bf16*)alloc((size_t)Dc * (Hc * Vc) * 2);
  __bf16* wob = (__bf16*)alloc((size_t)(Hc * Vc) * Dc * 2);
  float*  qp  = (float*)alloc((size_t)BT * (Hc * Kc) * 4);     // pre-conv proj
  float*  kp  = (float*)alloc((size_t)BT * (Hc * Kc) * 4);
  float*  vp  = (float*)alloc((size_t)BT * (Hc * Vc) * 4);
  float*  qn  = (float*)alloc((size_t)BT * (Hc * Kc) * 4);     // post conv+norm
  float*  kn  = (float*)alloc((size_t)BT * (Hc * Kc) * 4);
  float*  vn  = (float*)alloc((size_t)BT * (Hc * Vc) * 4);
  float*  gb  = (float*)alloc((size_t)BT * Hc * 4);
  float*  bb  = (float*)alloc((size_t)BT * Hc * 4);
  __bf16* obu = (__bf16*)alloc((size_t)BT * (Hc * Vc) * 2);    // normed o, bf16

  // 1) bf16 conversions (x and all GEMM weights, row-major, no transpose)
  {
    size_t n;
    n = (size_t)BT * Dc;
    cvt_f32_bf16<<<dim3((unsigned)((n + 255) / 256)), 256, 0, stream>>>(x, xb, n);
    n = (size_t)Dc * (Hc * Kc);
    cvt_f32_bf16<<<dim3((unsigned)((n + 255) / 256)), 256, 0, stream>>>(Wq, wqb, n);
    cvt_f32_bf16<<<dim3((unsigned)((n + 255) / 256)), 256, 0, stream>>>(Wk, wkb, n);
    n = (size_t)Dc * (Hc * Vc);
    cvt_f32_bf16<<<dim3((unsigned)((n + 255) / 256)), 256, 0, stream>>>(Wv, wvb, n);
    n = (size_t)(Hc * Vc) * Dc;
    cvt_f32_bf16<<<dim3((unsigned)((n + 255) / 256)), 256, 0, stream>>>(Wo, wob, n);
  }

  // 2) q/k/v projections via WMMA bf16 GEMM (TDM-staged B panels)
  wmma_gemm_bf16<<<dim3(BT / 64, (Hc * Kc) / 64), 128, 0, stream>>>(
      xb, wqb, qp, BT, Hc * Kc, Dc);
  wmma_gemm_bf16<<<dim3(BT / 64, (Hc * Kc) / 64), 128, 0, stream>>>(
      xb, wkb, kp, BT, Hc * Kc, Dc);
  wmma_gemm_bf16<<<dim3(BT / 64, (Hc * Vc) / 64), 128, 0, stream>>>(
      xb, wvb, vp, BT, Hc * Vc, Dc);

  // 3) short conv + SiLU (+ L2 norm for q,k; q scaled by K^-0.5)
  conv_silu_l2norm_qk<<<BT * Hc, 32, 0, stream>>>(qp, cqw, qn, 0.08838834764831843f);
  conv_silu_l2norm_qk<<<BT * Hc, 32, 0, stream>>>(kp, ckw, kn, 1.0f);
  {
    size_t nv = (size_t)BT * (Hc * Vc);
    conv_silu_v<<<dim3((unsigned)((nv + 255) / 256)), 256, 0, stream>>>(vp, cvw, vn);
  }

  // 4) gate / beta
  beta_g_kernel<<<BT, 32, 0, stream>>>(x, Wb, Wa, A_log, dtb, bb, gb);

  // 5) gated delta-rule scan with fused RMSNorm (bf16 out)
  {
    size_t shmem = (size_t)(Kc * Vc + Kc + Kc + Vc) * sizeof(float); // ~133 KB
    gdn_scan_kernel<<<Bc * Hc, 1024, shmem, stream>>>(qn, kn, vn, gb, bb, norm_w, obu);
  }

  // 6) output projection via WMMA bf16 GEMM -> fp32 out
  wmma_gemm_bf16<<<dim3(BT / 16, Dc / 64), 128, 0, stream>>>(
      obu, wob, out, BT, Dc, Hc * Vc);
}
